// GSAM_2370821947402
// MI455X (gfx1250) — matrix operand order repaired and checked by reading, not compiled, not run on hardware
//
#include <hip/hip_runtime.h>
#include <cstdint>

// ---------- types ----------
typedef __bf16 bf16;
typedef __attribute__((ext_vector_type(8)))  __bf16 v8bf;
typedef __attribute__((ext_vector_type(16))) __bf16 v16bf;
typedef __attribute__((ext_vector_type(8)))  float  v8f;

static __device__ __forceinline__ v16bf cat16(v8bf lo, v8bf hi) {
    return __builtin_shufflevector(lo, hi, 0,1,2,3,4,5,6,7,8,9,10,11,12,13,14,15);
}

union U16B { uint4 u; __bf16 h[8]; };

// ---------- problem sizes ----------
#define BB 8
#define NN 2048
#define PP 512
#define DD 64

// async copy global -> LDS (128b per lane), tracked by ASYNCcnt
#define GASYNC_B128(lds_u32, gp) \
    asm volatile("global_load_async_to_lds_b128 %0, %1, off" :: "v"(lds_u32), "v"(gp) : "memory")

static __device__ __forceinline__ void wait_async0() {
#if __has_builtin(__builtin_amdgcn_s_wait_asynccnt)
    __builtin_amdgcn_s_wait_asynccnt(0);
#else
    asm volatile("s_wait_asynccnt 0" ::: "memory");
#endif
}

static __device__ __forceinline__ uint32_t lds_addr32(const void* p) {
    return (uint32_t)(uintptr_t)p;   // low 32 bits of generic ptr = LDS offset
}

// ---------------------------------------------------------------------------
// f32 -> bf16 convert (for weights)
// ---------------------------------------------------------------------------
__global__ void cvt_bf16_kernel(const float* __restrict__ in, bf16* __restrict__ out, int n) {
    int i = blockIdx.x * 256 + threadIdx.x;
    if (i < n) out[i] = (bf16)in[i];
}

// ---------------------------------------------------------------------------
// LayerNorm over last dim (512), write bf16. One block (256 thr) per row.
// var = E[x^2]-mu^2 (population, no eps) to match jnp.var / reference.
// ---------------------------------------------------------------------------
__global__ void ln_kernel(const float* __restrict__ x,
                          const float* __restrict__ w,
                          const float* __restrict__ b,
                          bf16* __restrict__ xn) {
    const int row = blockIdx.x;
    const int t   = threadIdx.x;
    const float* xr = x + (size_t)row * PP;
    float v0 = xr[t];
    float v1 = xr[t + 256];
    __shared__ float s1[256];
    __shared__ float s2[256];
    s1[t] = v0 + v1;
    s2[t] = v0 * v0 + v1 * v1;
    __syncthreads();
    for (int off = 128; off > 0; off >>= 1) {
        if (t < off) { s1[t] += s1[t + off]; s2[t] += s2[t + off]; }
        __syncthreads();
    }
    const float mu   = s1[0] * (1.0f / PP);
    const float var  = s2[0] * (1.0f / PP) - mu * mu;
    const float rstd = rsqrtf(var);
    bf16* xo = xn + (size_t)row * PP;
    xo[t]       = (bf16)((v0 - mu) * rstd * w[t]       + b[t]);
    xo[t + 256] = (bf16)((v1 - mu) * rstd * w[t + 256] + b[t + 256]);
}

// ---------------------------------------------------------------------------
// Tiled bf16 GEMM, fp32 WMMA accumulation (v_wmma_f32_16x16x32_bf16).
// Tile: BM=256, BN=64, BK=32; 256 threads = 8 waves; each wave computes a
// 32(M) x 64(N) strip = 2x4 accumulators (8 WMMA per K-step).
// A tile (and B tile when !BKM) staged with GLOBAL_LOAD_ASYNC_TO_LDS_B128,
// double-buffered, one s_barrier per K-step.
// B is N-major (N x K) when !BKM; K-major (K x N) when BKM (Z @ v).
// ---------------------------------------------------------------------------
template <int EPI, bool BKM>
__global__ __launch_bounds__(256)
void gemm_k(const bf16* __restrict__ A, long long sAb, int lda,
            const bf16* __restrict__ B, long long sBb, int ldb,
            int K,
            const float* __restrict__ f0, const float* __restrict__ f1,
            const float* __restrict__ f2, const float* __restrict__ f3,
            const bf16* __restrict__ h0, long long sh0, int ldh0,
            float* __restrict__ of, long long sof, int ldof,
            bf16* __restrict__ ob, long long sob, int ldob,
            bf16* __restrict__ ob2,
            float scale)
{
    // rows padded to 40 bf16 (80B) -> conflict-free b128 fragment loads
    __shared__ __align__(16) bf16 sA[2][256 * 40];
    __shared__ __align__(16) bf16 sB[2][64 * 40];

    const int t  = threadIdx.x;
    const int m0 = blockIdx.y * 256;
    const int n0 = blockIdx.x * 64;
    const int z  = blockIdx.z;

    A += (long long)z * sAb;
    B += (long long)z * sBb;

    const int wave = t >> 5;
    const int lane = t & 31;
    const int lh   = lane >> 4;   // half-select (ISA 16-bit A/B lane split)
    const int lm   = lane & 15;
    const int mW   = wave * 32;

    v8f vz = {0.f,0.f,0.f,0.f,0.f,0.f,0.f,0.f};
    v8f acc[2][4] = {{vz,vz,vz,vz},{vz,vz,vz,vz}};

    // global->LDS chunk mappings (16B chunks)
    const int ar0 = t >> 2;          // A rows 0..63 (+64/+128/+192)
    const int ak  = (t & 3) * 8;     // A k-chunk within BK
    const int bn  = t >> 2;          // B (N-major): n row 0..63
    const int bk  = (t & 3) * 8;
    const int tk  = t >> 3;          // B (K-major): k row 0..31
    const int tn  = (t & 7) * 8;     // 8 consecutive n

    const bf16* aBase = A + (size_t)(m0 + ar0) * lda + ak;
    const bf16* bBase = BKM ? (B + (size_t)tk * ldb + n0 + tn)
                            : (B + (size_t)(n0 + bn) * ldb + bk);

    // issue one K-step's async loads into buffer `buf`
    auto issue_async = [&](int buf, int k0) {
#pragma unroll
        for (int rb = 0; rb < 4; rb++) {
            uint32_t la = lds_addr32(&sA[buf][(ar0 + 64 * rb) * 40 + ak]);
            const bf16* gp = aBase + (size_t)(64 * rb) * lda + k0;
            GASYNC_B128(la, gp);
        }
        if constexpr (!BKM) {
            uint32_t lb = lds_addr32(&sB[buf][bn * 40 + bk]);
            const bf16* gp = bBase + k0;
            GASYNC_B128(lb, gp);
        }
    };

    uint4 bstage = {};                 // BKM: staged B data for next K-step
    issue_async(0, 0);
    if constexpr (BKM) bstage = *(const uint4*)(bBase);

    for (int k0 = 0, kt = 0; k0 < K; k0 += 32, ++kt) {
        const int cur = kt & 1;
        if constexpr (BKM) {           // transpose-scatter staged B into LDS
            U16B u; u.u = bstage;
#pragma unroll
            for (int i = 0; i < 8; i++) sB[cur][(tn + i) * 40 + tk] = u.h[i];
        }
        wait_async0();                 // own async loads of tile kt landed
        __syncthreads();               // all waves: tile kt visible; compute kt-1 done
        if (k0 + 32 < K) {
            issue_async(cur ^ 1, k0 + 32);
            if constexpr (BKM)
                bstage = *(const uint4*)(bBase + (size_t)(k0 + 32) * ldb);
        }

        // A fragments: lane(0..15)=row, K{0..7,16..23}; lane(16..31) K{8..15,24..31}
        v16bf afrag[2];
#pragma unroll
        for (int af = 0; af < 2; af++) {
            const bf16* arow = &sA[cur][(mW + af * 16 + lm) * 40];
            afrag[af] = cat16(*(const v8bf*)(arow + lh * 8),
                              *(const v8bf*)(arow + 16 + lh * 8));
        }
#pragma unroll
        for (int nf = 0; nf < 4; nf++) {
            const bf16* brow = &sB[cur][(nf * 16 + lm) * 40 + lh * 8];
            v16bf bfrag = cat16(*(const v8bf*)(brow), *(const v8bf*)(brow + 16));
#pragma unroll
            for (int af = 0; af < 2; af++) {
                acc[af][nf] = __builtin_amdgcn_wmma_f32_16x16x32_bf16(
                    false, afrag[af], false, bfrag, (short)0, acc[af][nf], false, false);
            }
        }
    }

    // epilogue: frag elem r -> lanes0-15: M=r, lanes16-31: M=8+r; N=lm
#pragma unroll
    for (int af = 0; af < 2; af++) {
#pragma unroll
        for (int nf = 0; nf < 4; nf++) {
#pragma unroll
            for (int r = 0; r < 8; r++) {
                const int row = m0 + mW + af * 16 + lh * 8 + r;   // per-batch row
                const int col = n0 + nf * 16 + lm;
                float v = acc[af][nf][r];
                if constexpr (EPI == 0) {
                    float tt = fmaxf(v + f0[col], 0.0f);
                    ob[(long long)z * sob + (size_t)row * ldob + col] = (bf16)tt;
                } else if constexpr (EPI == 1) {
                    float tt = fmaxf(v + f0[col], 0.0f);
                    float q  = tt * f1[col]      + f2[col];
                    float kk = tt * f1[64 + col] + f2[64 + col];
                    ob [(size_t)row * ldob + col] = (bf16)q;
                    ob2[(size_t)row * ldob + col] = (bf16)kk;
                } else if constexpr (EPI == 2) {
                    float tt = fmaxf(v * scale + f3[(size_t)row * ldof + col], 0.0f);
                    float zz = tt * tt * (1.0f / (float)NN);
                    of[(long long)z * sof + (size_t)row * ldof + col] = zz;
                    ob[(long long)z * sob + (size_t)row * ldob + col] = (bf16)zz;
                } else if constexpr (EPI == 3) {
                    float pv = (float)h0[(long long)z * sh0 + (size_t)row * ldh0 + col];
                    ob[(long long)z * sob + (size_t)row * ldob + col] = (bf16)(pv * v);
                } else { // EPI == 4
                    of[(size_t)row * ldof + col] = v + f0[col] + f3[(size_t)row * ldof + col];
                }
            }
        }
    }
}

// ---------------------------------------------------------------------------
extern "C" void kernel_launch(void* const* d_in, const int* in_sizes, int n_in,
                              void* d_out, int out_size, void* d_ws, size_t ws_size,
                              hipStream_t stream) {
    (void)in_sizes; (void)n_in; (void)out_size; (void)ws_size;

    const float* x    = (const float*)d_in[0];   // (8,2048,512)
    const float* ln_w = (const float*)d_in[1];
    const float* ln_b = (const float*)d_in[2];
    const float* gmm  = (const float*)d_in[3];   // (2,64)
    const float* bet  = (const float*)d_in[4];
    const float* bmat = (const float*)d_in[5];   // (2048,2048)
    const float* w1w  = (const float*)d_in[6];   // (1088,512)
    const float* w1b  = (const float*)d_in[7];
    const float* w2w  = (const float*)d_in[8];   // (512,512)
    const float* w2b  = (const float*)d_in[9];

    float* outF = (float*)d_out;                          // (8,2048,512)
    float* outZ = outF + (long long)BB * NN * PP;         // (8,2048,2048)

    // workspace carve-up (all bf16)
    char* w = (char*)d_ws;
    bf16* xn   = (bf16*)w; w += (size_t)BB * NN * PP * 2;     // reused as p*(Z@v)
    bf16* w1bf = (bf16*)w; w += (size_t)(2*PP + DD) * PP * 2;
    bf16* w2bf = (bf16*)w; w += (size_t)PP * PP * 2;
    bf16* pbuf = (bf16*)w; w += (size_t)BB * NN * PP * 2;
    bf16* vbuf = (bf16*)w; w += (size_t)BB * NN * PP * 2;
    bf16* qbuf = (bf16*)w; w += (size_t)BB * NN * DD * 2;
    bf16* kbuf = (bf16*)w; w += (size_t)BB * NN * DD * 2;
    bf16* zbuf = (bf16*)w; w += (size_t)BB * NN * NN * 2;
    bf16* opre = xn;

    const int MR = BB * NN;           // 16384 rows total

    cvt_bf16_kernel<<<((2*PP+DD)*PP + 255) / 256, 256, 0, stream>>>(w1w, w1bf, (2*PP+DD)*PP);
    cvt_bf16_kernel<<<(PP*PP + 255) / 256, 256, 0, stream>>>(w2w, w2bf, PP*PP);
    ln_kernel<<<MR, 256, 0, stream>>>(x, ln_w, ln_b, xn);

    // GEMM1 p: cols 0..511
    gemm_k<0, false><<<dim3(PP/64, MR/256, 1), 256, 0, stream>>>(
        xn, 0, PP, w1bf, 0, PP, PP,
        w1b, nullptr, nullptr, nullptr,
        nullptr, 0, 0, nullptr, 0, 0,
        pbuf, 0, PP, nullptr, 0.0f);
    // GEMM1 v: cols 512..1023
    gemm_k<0, false><<<dim3(PP/64, MR/256, 1), 256, 0, stream>>>(
        xn, 0, PP, w1bf + (size_t)PP * PP, 0, PP, PP,
        w1b + PP, nullptr, nullptr, nullptr,
        nullptr, 0, 0, nullptr, 0, 0,
        vbuf, 0, PP, nullptr, 0.0f);
    // GEMM1 o: cols 1024..1087 -> q,k via gamma/beta
    gemm_k<1, false><<<dim3(DD/64, MR/256, 1), 256, 0, stream>>>(
        xn, 0, PP, w1bf + (size_t)2 * PP * PP, 0, PP, PP,
        w1b + 2*PP, gmm, bet, nullptr,
        nullptr, 0, 0, nullptr, 0, 0,
        qbuf, 0, DD, kbuf, 0.0f);

    // qk = q @ k^T per batch; Z = relu(qk/8 + b)^2 / N
    gemm_k<2, false><<<dim3(NN/64, NN/256, BB), 256, 0, stream>>>(
        qbuf, (long long)NN * DD, DD, kbuf, (long long)NN * DD, DD, DD,
        nullptr, nullptr, nullptr, bmat,
        nullptr, 0, 0,
        outZ, (long long)NN * NN, NN,
        zbuf, (long long)NN * NN, NN, nullptr, 0.125f);

    // y = Z @ v per batch (B K-major); out_pre = p * y
    gemm_k<3, true><<<dim3(PP/64, NN/256, BB), 256, 0, stream>>>(
        zbuf, (long long)NN * NN, NN, vbuf, (long long)NN * PP, PP, NN,
        nullptr, nullptr, nullptr, nullptr,
        pbuf, (long long)NN * PP, PP,
        nullptr, 0, 0,
        opre, (long long)NN * PP, PP, nullptr, 0.0f);

    // out = out_pre @ w2^T + w2_b + x0
    gemm_k<4, false><<<dim3(PP/64, MR/256, 1), 256, 0, stream>>>(
        opre, 0, PP, w2bf, 0, PP, PP,
        w2b, nullptr, nullptr, x,
        nullptr, 0, 0,
        outF, 0, PP,
        nullptr, 0, 0, nullptr, 0.0f);
}
